// EGNNLayer_16587163698061
// MI455X (gfx1250) — compile-verified, matrix operand order checked by reading, not run on hardware
//
#include <hip/hip_runtime.h>

typedef __attribute__((ext_vector_type(16))) __bf16 v16bf;
typedef __attribute__((ext_vector_type(8)))  float  v8f;

#define FDIM 128
#define HDIM 128
#define TE   16          // edges / nodes per block tile
#define A1_K 256         // layer-1 K after folding dist2 into the C init
#define A1_STRIDE 264    // padded LDS stride in halves (breaks bank conflicts)
#define H_STRIDE  136

__device__ __forceinline__ float silu(float v) {
  return v / (1.0f + __expf(-v));
}

// Load a 16x32 bf16 WMMA fragment (A or B) given a row/col-major base.
// Element layout per CDNA5 ISA 7.12.2: lane holds index = lane&15,
// elements 0..7 = K (kbase + 8*(lane>>4)) .. +7, elements 8..15 = +16.
template<int STRIDE>
__device__ __forceinline__ v16bf load_frag(const __bf16* __restrict__ base,
                                           int idx, int kbase, int lane) {
  const int koff = (lane >> 4) << 3;
  const __bf16* p = base + idx * STRIDE + kbase + koff;
  v16bf r;
#pragma unroll
  for (int i = 0; i < 8; ++i) r[i] = p[i];
#pragma unroll
  for (int i = 0; i < 8; ++i) r[i + 8] = p[16 + i];
  return r;
}

// ---------------- init: zero agg, seed pos_out with pos ----------------
__global__ void egnn_init_kernel(const float* __restrict__ pos,
                                 float* __restrict__ agg,
                                 float* __restrict__ pos_out, int Nn) {
  const long i = (long)blockIdx.x * blockDim.x + threadIdx.x;
  if (i < (long)Nn * FDIM) agg[i] = 0.0f;
  if (i < (long)Nn * 3)    pos_out[i] = pos[i];
}

// ---------------- prep: bf16 transposed weight copies ----------------
__global__ void egnn_prep_kernel(const float* __restrict__ We1,
                                 const float* __restrict__ We2,
                                 const float* __restrict__ Wn,
                                 __bf16* __restrict__ w1T,
                                 __bf16* __restrict__ w2T,
                                 __bf16* __restrict__ wnT,
                                 float*  __restrict__ w1last) {
  const int i = blockIdx.x * blockDim.x + threadIdx.x;
  if (i < 128 * 256) {                       // We1[0:256][128] -> w1T[n][k]
    const int n = i >> 8, k = i & 255;
    w1T[i] = (__bf16)We1[k * 128 + n];
  } else if (i < 128 * 256 + 128 * 128) {    // We2 -> w2T[n][k]
    const int j = i - 32768, n = j >> 7, k = j & 127;
    w2T[j] = (__bf16)We2[k * 128 + n];
  } else if (i < 128 * 256 + 128 * 128 + 128 * 256) {  // Wn -> wnT[n][k]
    const int j = i - 49152, n = j >> 8, k = j & 255;
    wnT[j] = (__bf16)Wn[k * 128 + n];
  } else if (i < 81920 + 128) {              // dist2 row of We1
    const int n = i - 81920;
    w1last[n] = We1[256 * 128 + n];
  }
}

// ---------------- fused edge kernel ----------------
__global__ __launch_bounds__(256) void egnn_edge_kernel(
    const float* __restrict__ x, const float* __restrict__ pos,
    const int* __restrict__ eidx,
    const __bf16* __restrict__ w1T, const float* __restrict__ w1last,
    const float* __restrict__ be1,
    const __bf16* __restrict__ w2T, const float* __restrict__ be2,
    const float* __restrict__ Wc, const float* __restrict__ bc,
    float* __restrict__ agg, float* __restrict__ pos_out, int E) {
  __shared__ __bf16 aT[TE * A1_STRIDE];
  __shared__ __bf16 hT[TE * H_STRIDE];
  __shared__ float  dist2s[TE];
  __shared__ float  rels[TE][3];
  __shared__ int    rowi[TE];
  __shared__ float  cw[TE];

  const int  t     = threadIdx.x;
  const int  lane  = t & 31;
  const int  wv    = t >> 5;                // wave id -> 16-column slice
  const long ebase = (long)blockIdx.x * TE;
  if (ebase >= E) return;

  if (t < TE) {
    const long ge = ebase + t;
    const long e  = ge < E ? ge : (long)E - 1;
    const int r = eidx[e];
    const int c = eidx[(long)E + e];
    rowi[t] = (ge < E) ? r : -1;
    const float dx = pos[3 * r + 0] - pos[3 * c + 0];
    const float dy = pos[3 * r + 1] - pos[3 * c + 1];
    const float dz = pos[3 * r + 2] - pos[3 * c + 2];
    rels[t][0] = dx; rels[t][1] = dy; rels[t][2] = dz;
    dist2s[t] = dx * dx + dy * dy + dz * dz;
    cw[t] = 0.0f;
  }

  { // gather [x_row | x_col] -> bf16 LDS tile (16 features per thread)
    const int  e   = t >> 4;
    const int  seg = t & 15;
    long ge = ebase + e; if (ge >= E) ge = (long)E - 1;
    const int  node = eidx[(seg < 8 ? 0 : (long)E) + ge];
    const int  f0   = (seg & 7) << 4;
    const float* sp = x + (long)node * FDIM + f0;
    __bf16* dp = &aT[e * A1_STRIDE + (seg < 8 ? 0 : FDIM) + f0];
#pragma unroll
    for (int i = 0; i < 16; ++i) dp[i] = (__bf16)sp[i];
  }
  __syncthreads();

  const int ncol = (wv << 4) + (lane & 15);  // output feature 0..127
  const int mhi  = (lane >> 4) << 3;         // +8 row offset, upper half-wave

  // layer 1: h = silu([x_r|x_c] @ We1[0:256] + be1 + dist2 * We1[256])
  v8f acc;
  {
    const float bw = be1[ncol];
    const float wl = w1last[ncol];
#pragma unroll
    for (int v = 0; v < 8; ++v) acc[v] = bw + dist2s[v + mhi] * wl;
  }
#pragma unroll
  for (int ks = 0; ks < A1_K / 32; ++ks) {
    v16bf a = load_frag<A1_STRIDE>(aT, lane & 15, ks * 32, lane);
    v16bf b = load_frag<A1_K>(w1T + (long)(wv << 4) * A1_K, lane & 15, ks * 32, lane);
    acc = __builtin_amdgcn_wmma_f32_16x16x32_bf16(false, a, false, b,
                                                  (short)0, acc, false, false);
  }
#pragma unroll
  for (int v = 0; v < 8; ++v)
    hT[(v + mhi) * H_STRIDE + ncol] = (__bf16)silu(acc[v]);
  __syncthreads();

  // layer 2: m = silu(h @ We2 + be2)
  v8f acc2;
  {
    const float bw = be2[ncol];
#pragma unroll
    for (int v = 0; v < 8; ++v) acc2[v] = bw;
  }
#pragma unroll
  for (int ks = 0; ks < HDIM / 32; ++ks) {
    v16bf a = load_frag<H_STRIDE>(hT, lane & 15, ks * 32, lane);
    v16bf b = load_frag<HDIM>(w2T + (long)(wv << 4) * HDIM, lane & 15, ks * 32, lane);
    acc2 = __builtin_amdgcn_wmma_f32_16x16x32_bf16(false, a, false, b,
                                                   (short)0, acc2, false, false);
  }

  const float wcn = Wc[ncol];
#pragma unroll
  for (int v = 0; v < 8; ++v) {
    const int M = v + mhi;
    const int r = rowi[M];
    const float mv = silu(acc2[v]);
    if (r >= 0) {
      atomicAdd(&agg[(long)r * HDIM + ncol], mv);  // segment_sum(m_ij)
      atomicAdd(&cw[M], mv * wcn);                 // m_ij . Wc (LDS reduce)
    }
  }
  __syncthreads();

  if (t < TE * 3) {
    const int e = t / 3, c = t % 3;
    const int r = rowi[e];
    if (r >= 0) {
      const float w = tanhf(cw[e] + bc[0]);
      atomicAdd(&pos_out[(long)r * 3 + c], rels[e][c] * w);
    }
  }
}

// ---------------- node kernel: x_out = silu([x | agg] @ Wn + bn) ----------------
__global__ __launch_bounds__(256) void egnn_node_kernel(
    const float* __restrict__ x, const float* __restrict__ agg,
    const __bf16* __restrict__ wnT, const float* __restrict__ bn,
    float* __restrict__ x_out, int Nn) {
  __shared__ __bf16 aT[TE * A1_STRIDE];
  const int  t     = threadIdx.x;
  const int  lane  = t & 31;
  const int  wv    = t >> 5;
  const long nbase = (long)blockIdx.x * TE;
  if (nbase >= Nn) return;

  {
    const int nloc = t >> 4, seg = t & 15;
    long node = nbase + nloc; if (node >= Nn) node = Nn - 1;
    const int f0 = (seg & 7) << 4;
    const float* sp = (seg < 8 ? x : agg) + node * FDIM + f0;
    __bf16* dp = &aT[nloc * A1_STRIDE + (seg < 8 ? 0 : FDIM) + f0];
#pragma unroll
    for (int i = 0; i < 16; ++i) dp[i] = (__bf16)sp[i];
  }
  __syncthreads();

  const int ncol = (wv << 4) + (lane & 15);
  const int mhi  = (lane >> 4) << 3;
  v8f acc;
  {
    const float bw = bn[ncol];
#pragma unroll
    for (int v = 0; v < 8; ++v) acc[v] = bw;
  }
#pragma unroll
  for (int ks = 0; ks < A1_K / 32; ++ks) {
    v16bf a = load_frag<A1_STRIDE>(aT, lane & 15, ks * 32, lane);
    v16bf b = load_frag<A1_K>(wnT + (long)(wv << 4) * A1_K, lane & 15, ks * 32, lane);
    acc = __builtin_amdgcn_wmma_f32_16x16x32_bf16(false, a, false, b,
                                                  (short)0, acc, false, false);
  }
#pragma unroll
  for (int v = 0; v < 8; ++v) {
    const long row = nbase + v + mhi;
    if (row < Nn) x_out[row * FDIM + ncol] = silu(acc[v]);
  }
}

extern "C" void kernel_launch(void* const* d_in, const int* in_sizes, int n_in,
                              void* d_out, int out_size, void* d_ws, size_t ws_size,
                              hipStream_t stream) {
  const float* x    = (const float*)d_in[0];
  const float* pos  = (const float*)d_in[1];
  const int*   eidx = (const int*)d_in[2];
  const float* We1  = (const float*)d_in[3];
  const float* be1  = (const float*)d_in[4];
  const float* We2  = (const float*)d_in[5];
  const float* be2  = (const float*)d_in[6];
  const float* Wn   = (const float*)d_in[7];
  const float* bn   = (const float*)d_in[8];
  const float* Wc   = (const float*)d_in[9];
  const float* bc   = (const float*)d_in[10];

  const int Nn = in_sizes[0] / FDIM;   // 20000
  const int E  = in_sizes[2] / 2;      // 640000

  float* out     = (float*)d_out;
  float* x_out   = out;
  float* pos_out = out + (size_t)Nn * FDIM;

  // workspace carve-up
  char* ws = (char*)d_ws;
  size_t off = 0;
  float* agg = (float*)(ws + off);      off += (size_t)Nn * FDIM * sizeof(float);
  __bf16* w1T = (__bf16*)(ws + off);    off += (size_t)128 * 256 * sizeof(__bf16);
  __bf16* w2T = (__bf16*)(ws + off);    off += (size_t)128 * 128 * sizeof(__bf16);
  __bf16* wnT = (__bf16*)(ws + off);    off += (size_t)128 * 256 * sizeof(__bf16);
  float* w1last = (float*)(ws + off);   off += 128 * sizeof(float);

  const int initN = Nn * FDIM;
  egnn_init_kernel<<<(initN + 255) / 256, 256, 0, stream>>>(pos, agg, pos_out, Nn);

  const int prepN = 128 * 256 + 128 * 128 + 128 * 256 + 128;
  egnn_prep_kernel<<<(prepN + 255) / 256, 256, 0, stream>>>(We1, We2, Wn,
                                                            w1T, w2T, wnT, w1last);

  egnn_edge_kernel<<<(E + TE - 1) / TE, 256, 0, stream>>>(
      x, pos, eidx, w1T, w1last, be1, w2T, be2, Wc, bc, agg, pos_out, E);

  egnn_node_kernel<<<(Nn + TE - 1) / TE, 256, 0, stream>>>(
      x, agg, wnT, bn, x_out, Nn);
}
